// GraphConvolutionImprove_43559558316212
// MI455X (gfx1250) — matrix operand order repaired and checked by reading, not compiled, not run on hardware
//
#include <hip/hip_runtime.h>

typedef __attribute__((ext_vector_type(16))) __bf16 v16bf;
typedef __attribute__((ext_vector_type(8)))  __bf16 v8bf;
typedef __attribute__((ext_vector_type(4)))  __bf16 v4bf;
typedef __attribute__((ext_vector_type(8)))  float  v8f;
typedef __attribute__((ext_vector_type(4)))  int    v4i;

#define FIN   64
#define FOUT  64
#define KNB   9
#define KDIM  (KNB * FIN)      // 576
#define MT    64               // m-rows per workgroup
#define LDA   584              // padded LDS stride (elems): 1168 B, 16B-aligned, conflict-free
#define LDB   584
#define KB    32               // K per WMMA
#define NKB   (KDIM / KB)      // 18

// gfx1250 async global->LDS copy (ASYNCcnt-tracked), with safe fallback.
#if __has_builtin(__builtin_amdgcn_global_load_async_to_lds_b128)
#define HAVE_ASYNC_LDS 1
typedef __attribute__((address_space(1))) v4i* as1_v4i_p;
typedef __attribute__((address_space(3))) v4i* as3_v4i_p;
#else
#define HAVE_ASYNC_LDS 0
#endif

__device__ __forceinline__ v16bf ldfrag(const __bf16* p, int hi_off) {
    v8bf lo = *(const v8bf*)p;
    v8bf hi = *(const v8bf*)(p + hi_off);
    return __builtin_shufflevector(lo, hi,
        0,1,2,3,4,5,6,7,8,9,10,11,12,13,14,15);
}

__device__ __forceinline__ v8f wmma_bf16(v16bf a, v16bf b, v8f c) {
    return __builtin_amdgcn_wmma_f32_16x16x32_bf16(
        false, a, false, b, (short)0, c, false, false);
}

// ---- pre-kernel: weight fp32 (64x576) -> bf16 in workspace, once per launch ----
__global__ __launch_bounds__(256)
void wcvt_kernel(const float* __restrict__ w, __bf16* __restrict__ wb) {
    int t = blockIdx.x * 256 + threadIdx.x;          // 0..9215 float4 chunks
    float4 v = ((const float4*)w)[t];
    v4bf pk = { (__bf16)v.x, (__bf16)v.y, (__bf16)v.z, (__bf16)v.w };
    ((v4bf*)wb)[t] = pk;
}

// ---- main fused gather + GEMM + bias + ELU ----
__global__ __launch_bounds__(256)
void gconv_wmma_bf16(const float*  __restrict__ x,
                     const int*    __restrict__ idx,
                     const __bf16* __restrict__ wb,
                     const float*  __restrict__ bias,
                     float*        __restrict__ out,
                     int Mrows)
{
    __shared__ __align__(16) __bf16 Alds[MT   * LDA];   // 74,752 B
    __shared__ __align__(16) __bf16 Blds[FOUT * LDB];   // 74,752 B
    __shared__ int Ilds[MT * KNB];                      //  2,304 B  (~152 KB: 2 WGs/WGP)

    const int tid = threadIdx.x;
    const int nb  = blockIdx.y;
    const int m0  = blockIdx.x * MT;

    // ---- Stage indices: 576 contiguous ints starting at idx + m0*9 ----
    const int idx_limit = Mrows * KNB - m0 * KNB;       // valid count
#pragma unroll
    for (int i = 0; i < 3; ++i) {
        int t = tid + i * 256;
        if (t < MT * KNB)
            Ilds[t] = (t < idx_limit) ? idx[m0 * KNB + t] : 0;
    }

    // ---- Stage weight (bf16, pre-converted) -> Blds: 64 rows x 72 16B chunks ----
#pragma unroll 2
    for (int i = 0; i < 18; ++i) {
        int c   = tid + i * 256;        // 0..4607
        int row = c / 72;
        int j   = c - row * 72;         // 16B chunk within row
#if HAVE_ASYNC_LDS
        __builtin_amdgcn_global_load_async_to_lds_b128(
            (as1_v4i_p)(wb + row * KDIM + j * 8),
            (as3_v4i_p)(&Blds[row * LDB + j * 8]),
            0, 0);
#else
        *(uint4*)&Blds[row * LDB + j * 8] =
            *(const uint4*)(wb + row * KDIM + j * 8);
#endif
    }

    // ---- Gather x -> Alds (bf16): 64 rows x 9 chunks, 16 threads per 64-float chunk ----
#pragma unroll 2
    for (int it = 0; it < 36; ++it) {
        int chunk  = it * 16 + (tid >> 4);   // 0..575
        int lane16 = tid & 15;
        int row    = chunk / KNB;
        int k      = chunk - row * KNB;
        int g      = Ilds[chunk];            // 0 for tail rows (safe)
        float4 v = *(const float4*)(x + (((size_t)nb * Mrows + g) * FIN) + lane16 * 4);
        v4bf pk = { (__bf16)v.x, (__bf16)v.y, (__bf16)v.z, (__bf16)v.w };
        *(v4bf*)&Alds[row * LDA + k * FIN + lane16 * 4] = pk;
    }

#if HAVE_ASYNC_LDS
#if __has_builtin(__builtin_amdgcn_s_wait_asynccnt)
    __builtin_amdgcn_s_wait_asynccnt(0);
#else
    asm volatile("s_wait_asynccnt 0x0" ::: "memory");
#endif
#endif
    __syncthreads();

    // ---- WMMA: 8 waves; wave computes 16(m) x 32(f): 1 A-frag, 2 B-frags / kb ----
    const int wave  = tid >> 5;
    const int lane  = tid & 31;
    const int mq    = wave & 3;    // 0..3 : 16-row m-subtile
    const int nq    = wave >> 2;   // 0..1 : 32-col f-slot
    const int l15   = lane & 15;
    const int khalf = lane >> 4;

    const __bf16* aBase  = &Alds[(mq * 16 + l15) * LDA + khalf * 16];
    const __bf16* bBase0 = &Blds[(nq * 32 + l15) * LDB + khalf * 16];
    const __bf16* bBase1 = bBase0 + 16 * LDB;

    v8f acc0 = {};
    v8f acc1 = {};
#pragma unroll
    for (int kb = 0; kb < NKB; ++kb) {
        v16bf a  = ldfrag(aBase  + kb * KB, 8);   // A 16x32: 16 contiguous K
        v16bf b0 = ldfrag(bBase0 + kb * KB, 8);   // B 32x16: 16 contiguous K
        v16bf b1 = ldfrag(bBase1 + kb * KB, 8);
        acc0 = wmma_bf16(a, b0, acc0);
        acc1 = wmma_bf16(a, b1, acc1);
    }

    // ---- Epilogue: bias + ELU + NT store ----
    const int   col0  = nq * 32 + l15;
    const int   col1  = col0 + 16;
    const float bv0   = bias[col0];
    const float bv1   = bias[col1];
    const int   mbase = m0 + mq * 16 + khalf * 8;
#pragma unroll
    for (int v = 0; v < 8; ++v) {
        int m = mbase + v;
        if (m < Mrows) {
            float* rowp = out + (((size_t)nb * Mrows) + m) * FOUT;
            float y0 = acc0[v] + bv0;
            float y1 = acc1[v] + bv1;
            __builtin_nontemporal_store(y0 > 0.f ? y0 : (__expf(y0) - 1.f), rowp + col0);
            __builtin_nontemporal_store(y1 > 0.f ? y1 : (__expf(y1) - 1.f), rowp + col1);
        }
    }
}

extern "C" void kernel_launch(void* const* d_in, const int* in_sizes, int n_in,
                              void* d_out, int out_size, void* d_ws, size_t ws_size,
                              hipStream_t stream) {
    const float* x    = (const float*)d_in[0];
    const int*   idx  = (const int*)  d_in[1];
    const float* w    = (const float*)d_in[2];
    const float* bias = (const float*)d_in[3];
    float*       out  = (float*)d_out;
    __bf16*      wb   = (__bf16*)d_ws;              // 73,728 B of scratch

    int M = in_sizes[1] / KNB;                      // index_list is (M, 9)
    int N = (int)(in_sizes[0] / ((size_t)M * FIN));

    wcvt_kernel<<<(FOUT * KDIM) / (256 * 4), 256, 0, stream>>>(w, wb);

    dim3 grid((M + MT - 1) / MT, N);
    gconv_wmma_bf16<<<grid, 256, 0, stream>>>(x, idx, wb, bias, out, M);
}
